// CombinedMSESSIMLoss_60988535603478
// MI455X (gfx1250) — matrix operand order repaired and checked by reading, not compiled, not run on hardware
//
#include <hip/hip_runtime.h>

// ---------------------------------------------------------------------------
// CombinedMSESSIMLoss for MI455X (gfx1250, wave32).
//   pass1: MSE sum + min/max(target)            (memory bound, float4 loads)
//   pass2: finalize scalars (mse, data_range)
//   pass3: SSIM via f16 WMMA  (Gaussian filter == banded GEMM, 8 wmma sites)
//   pass4: Sobel (incl. batch-axis [1,2,1] smoothing) -> Pearson partial sums
//   pass5: combine -> d_out[0]
// Deterministic: per-block partials in d_ws, fixed-order tree reductions,
// no atomics. All launches on `stream`.
// ---------------------------------------------------------------------------

#define NIMG   65536
#define IMH    28
#define IMHW   784
#define NVALID 18
#define NBLK   1024
#define IPB    64                    // images per block = NIMG/NBLK

#define STAT_BASE 16                 // 3 floats per block
#define SSIM_BASE (16 + 3*NBLK)      // 1 float per block
#define EPI_BASE  (SSIM_BASE + NBLK) // 5 floats per block

typedef __attribute__((ext_vector_type(16))) _Float16 v16h;
typedef __attribute__((ext_vector_type(8)))  float    v8f;

// 11-tap Gaussian, sigma=1.5, normalized (pytorch_msssim window)
__constant__ float c_gw[11] = {
  0.0010284f, 0.0075987f, 0.0360008f, 0.1093606f, 0.2130058f,
  0.2660117f,
  0.2130058f, 0.1093606f, 0.0360008f, 0.0075987f, 0.0010284f };

// banded filter matrix Gh[k, c] = g[k - c]  (VALID conv as GEMM)
__device__ __forceinline__ float gband(int c, int k) {
  int d = k - c;
  return (d >= 0 && d < 11 && k < IMH) ? c_gw[d] : 0.f;
}

__device__ __forceinline__ float blockReduceSum(float v, float* sm, int nthreads) {
  int tid = threadIdx.x;
  sm[tid] = v; __syncthreads();
  for (int s = 128; s > 0; s >>= 1) {
    if (tid < s && tid + s < nthreads) sm[tid] += sm[tid + s];
    __syncthreads();
  }
  float r = sm[0]; __syncthreads();
  return r;
}

// ---------------- pass 1: MSE sum + min/max(target) ------------------------
__global__ __launch_bounds__(256)
void k_stats(const float* __restrict__ o, const float* __restrict__ t,
             float* __restrict__ ws)
{
  __shared__ float sA[256], sB[256], sC[256];
  const long long n4 = (long long)NIMG * IMHW / 4;
  long long idx = (long long)blockIdx.x * blockDim.x + threadIdx.x;
  const long long stride = (long long)gridDim.x * blockDim.x;
  const float4* o4 = (const float4*)o;
  const float4* t4 = (const float4*)t;
  float ss = 0.f, mn = 3.4e38f, mx = -3.4e38f;
  for (long long i = idx; i < n4; i += stride) {
    float4 a = o4[i], b = t4[i];
    float d0 = a.x - b.x, d1 = a.y - b.y, d2 = a.z - b.z, d3 = a.w - b.w;
    ss += d0*d0 + d1*d1 + d2*d2 + d3*d3;
    mn = fminf(mn, fminf(fminf(b.x, b.y), fminf(b.z, b.w)));
    mx = fmaxf(mx, fmaxf(fmaxf(b.x, b.y), fmaxf(b.z, b.w)));
  }
  int tid = threadIdx.x;
  sA[tid] = ss; sB[tid] = mn; sC[tid] = mx; __syncthreads();
  for (int s = 128; s > 0; s >>= 1) {
    if (tid < s) {
      sA[tid] += sA[tid + s];
      sB[tid]  = fminf(sB[tid], sB[tid + s]);
      sC[tid]  = fmaxf(sC[tid], sC[tid + s]);
    }
    __syncthreads();
  }
  if (tid == 0) {
    ws[STAT_BASE + 3*blockIdx.x + 0] = sA[0];
    ws[STAT_BASE + 3*blockIdx.x + 1] = sB[0];
    ws[STAT_BASE + 3*blockIdx.x + 2] = sC[0];
  }
}

// ---------------- pass 2: finalize scalars ---------------------------------
__global__ __launch_bounds__(256)
void k_finalize_stats(float* __restrict__ ws)
{
  __shared__ float sA[256], sB[256], sC[256];
  int tid = threadIdx.x;
  float ss = 0.f, mn = 3.4e38f, mx = -3.4e38f;
  for (int i = tid; i < NBLK; i += 256) {
    ss += ws[STAT_BASE + 3*i + 0];
    mn  = fminf(mn, ws[STAT_BASE + 3*i + 1]);
    mx  = fmaxf(mx, ws[STAT_BASE + 3*i + 2]);
  }
  sA[tid] = ss; sB[tid] = mn; sC[tid] = mx; __syncthreads();
  for (int s = 128; s > 0; s >>= 1) {
    if (tid < s) {
      sA[tid] += sA[tid + s];
      sB[tid]  = fminf(sB[tid], sB[tid + s]);
      sC[tid]  = fmaxf(sC[tid], sC[tid + s]);
    }
    __syncthreads();
  }
  if (tid == 0) {
    float mse = sA[0] / (float)((long long)NIMG * IMHW);
    ws[0] = mse;           // mse (also drives PSNR)
    ws[1] = sB[0];         // min(target)
    ws[2] = sC[0];         // max(target)
    ws[3] = sC[0] - sB[0]; // data_range
  }
}

// ---------------- pass 3: SSIM via WMMA ------------------------------------
// One 160-thread block (5 waves) per IPB=64 images (looped); wave p owns map p of
// {x, y, x*x, y*y, x*y}, selected BRANCHLESSLY via scalarized 0/1 weights:
//   v = u*w,  u = x + selY1*(y-x),  w = selOne + selX2*x + selY2*y
// Each separable Gaussian pass = 2x2 tiles of v_wmma_f32_16x16x32_f16.
// Images staged zero-padded to 32x32 in LDS -> unguarded, mergeable ds loads.
__global__ __launch_bounds__(160)
void k_ssim(const float* __restrict__ o, const float* __restrict__ t,
            float* __restrict__ ws)
{
  __shared__ __align__(16) float    sX[32*32];    // zero-padded image (output)
  __shared__ __align__(16) float    sY[32*32];    // zero-padded image (target)
  __shared__ __align__(32) _Float16 sF[5][32*32]; // stage-1 output (f16)
  __shared__ __align__(16) float    sM[5][32*32]; // stage-2 output (f32)
  __shared__ float sRed[160];

  const int tid  = threadIdx.x;
  const int lane = tid & 31;
  const int half = lane >> 4;         // A half-lane split (ISA layout)
  const int mrow = lane & 15;
  const int wvs  = __builtin_amdgcn_readfirstlane(tid >> 5); // map id, SGPR

  // branchless map-selection weights (scalar per wave)
  const float selY1  = (wvs == 1 || wvs == 3) ? 1.f : 0.f;
  const float selOne = (wvs <= 1) ? 1.f : 0.f;
  const float selX2  = (wvs == 2) ? 1.f : 0.f;
  const float selY2  = (wvs >= 3) ? 1.f : 0.f;

  const float dr = ws[3];
  const float C1 = (0.01f * dr) * (0.01f * dr);
  const float C2 = (0.03f * dr) * (0.03f * dr);

  // zero-fill padded LDS images once (interior rewritten each image)
  for (int i = tid; i < 32*32; i += 160) { sX[i] = 0.f; sY[i] = 0.f; }

  // ---- loop-invariant Gaussian WMMA fragments (functions of lane only) ----
  v16h bG[2];   // stage-1 B operand: B[k][n=nt*16+mrow] = gband(n, k), k table order
  #pragma unroll
  for (int nt = 0; nt < 2; ++nt)
    #pragma unroll
    for (int e = 0; e < 16; ++e) {
      int j = e >> 1, lo = e & 1;
      int k = 2*j + ((j >= 4) ? 8 : 0) + 8*half + lo;
      bG[nt][e] = (_Float16)gband(nt*16 + mrow, k);
    }

  v16h aG[2];   // stage-2 A operand: A[m=mt*16+mrow][k] = gband(m, k)
  #pragma unroll
  for (int mt = 0; mt < 2; ++mt)
    #pragma unroll
    for (int e = 0; e < 16; ++e) {
      int j = e >> 1, lo = e & 1;
      int k = 2*j + ((j >= 4) ? 8 : 0) + 8*half + lo;
      aG[mt][e] = (_Float16)gband(mt*16 + mrow, k);
    }

  float ssimAcc = 0.f;

  for (int it = 0; it < IPB; ++it) {
    const int img = blockIdx.x * IPB + it;
    const float4* xo4 = (const float4*)(o + (size_t)img * IMHW);
    const float4* xt4 = (const float4*)(t + (size_t)img * IMHW);
    __syncthreads();
    // stage images: 196 float4s, rows of 28 = 7 aligned float4s, stride 32
    for (int q = tid; q < IMHW/4; q += 160) {
      int r  = q / 7;
      int c4 = (q - r*7) * 4;
      *(float4*)&sX[r*32 + c4] = xo4[q];
      *(float4*)&sY[r*32 + c4] = xt4[q];
    }
    __syncthreads();

    // ---- stage 1: Out1(32x32) = Map(32x32) * Gh(32x32) (horizontal pass)
    v8f acc1[2][2];
    #pragma unroll
    for (int mt = 0; mt < 2; ++mt) {
      const int row = mt*16 + mrow;
      v16h a;
      #pragma unroll
      for (int ep = 0; ep < 8; ++ep) {       // element pairs: adjacent k
        int kbase = 2*ep + ((ep >= 4) ? 8 : 0) + 8*half;
        float2 xv = *(const float2*)&sX[row*32 + kbase];
        float2 yv = *(const float2*)&sY[row*32 + kbase];
        float u0 = xv.x + selY1*(yv.x - xv.x);
        float w0 = selOne + selX2*xv.x + selY2*yv.x;
        float u1 = xv.y + selY1*(yv.y - xv.y);
        float w1 = selOne + selX2*xv.y + selY2*yv.y;
        a[2*ep]   = (_Float16)(u0*w0);
        a[2*ep+1] = (_Float16)(u1*w1);
      }
      #pragma unroll
      for (int nt = 0; nt < 2; ++nt) {
        v8f c = {};
        acc1[mt][nt] = __builtin_amdgcn_wmma_f32_16x16x32_f16(
            false, a, false, bG[nt], (short)0, c, false, false);
      }
    }
    // spill Out1 (f16) to LDS; D layout: row = r + 8*half, col = lane%16
    #pragma unroll
    for (int mt = 0; mt < 2; ++mt)
      #pragma unroll
      for (int nt = 0; nt < 2; ++nt)
        #pragma unroll
        for (int r = 0; r < 8; ++r)
          sF[wvs][(mt*16 + r + 8*half)*32 + nt*16 + mrow] =
              (_Float16)acc1[mt][nt][r];
    // same-wave LDS is in-order; stage 2 reads only this wave's region

    // ---- stage 2: Out2(32x32) = Gh^T(32x32) * Out1(32x32) (vertical pass)
    v8f acc2[2][2];
    #pragma unroll
    for (int mt = 0; mt < 2; ++mt) {
      #pragma unroll
      for (int nt = 0; nt < 2; ++nt) {
        // B: lane%16=N, reg=K (table order) -> column nt*16+mrow of Out1
        v16h b;
        #pragma unroll
        for (int e = 0; e < 16; ++e) {
          int j = e >> 1, lo = e & 1;
          int k = 2*j + ((j >= 4) ? 8 : 0) + 8*half + lo;
          b[e] = sF[wvs][k*32 + nt*16 + mrow];
        }
        v8f c = {};
        acc2[mt][nt] = __builtin_amdgcn_wmma_f32_16x16x32_f16(
            false, aG[mt], false, b, (short)0, c, false, false);
      }
    }
    #pragma unroll
    for (int mt = 0; mt < 2; ++mt)
      #pragma unroll
      for (int nt = 0; nt < 2; ++nt)
        #pragma unroll
        for (int r = 0; r < 8; ++r)
          sM[wvs][(mt*16 + r + 8*half)*32 + nt*16 + mrow] = acc2[mt][nt][r];
    __syncthreads();

    // ---- SSIM map over the 18x18 VALID region
    for (int px = tid; px < NVALID*NVALID; px += 160) {
      int i = px / NVALID, j2 = px - i*NVALID;
      int oo = i*32 + j2;
      float mu1 = sM[0][oo], mu2 = sM[1][oo];
      float exx = sM[2][oo], eyy = sM[3][oo], exy = sM[4][oo];
      float mu1s = mu1*mu1, mu2s = mu2*mu2, m12 = mu1*mu2;
      float s1 = exx - mu1s, s2 = eyy - mu2s, s12 = exy - m12;
      float num = (2.f*m12 + C1) * (2.f*s12 + C2);
      float den = (mu1s + mu2s + C1) * (s1 + s2 + C2);
      ssimAcc += num / den;
    }
  }
  float tot = blockReduceSum(ssimAcc, sRed, 160);
  if (tid == 0) ws[SSIM_BASE + blockIdx.x] = tot;
}

// ---------------- pass 4: Sobel + Pearson partial sums ---------------------
__global__ __launch_bounds__(256)
void k_epi(const float* __restrict__ o, const float* __restrict__ t,
           float* __restrict__ ws)
{
  __shared__ __align__(16) float sbx[IMHW], sby[IMHW];
  __shared__ float sRed[256];
  const int tid = threadIdx.x;
  float a1 = 0.f, a2 = 0.f, a3 = 0.f, a4 = 0.f, a5 = 0.f;

  for (int it = 0; it < IPB; ++it) {
    const int b  = blockIdx.x * IPB + it;
    const int bm = (b == 0) ? 0 : b - 1;               // symmetric pad in batch
    const int bp = (b == NIMG-1) ? (NIMG-1) : b + 1;
    const float4* om = (const float4*)(o + (size_t)bm * IMHW);
    const float4* oc = (const float4*)(o + (size_t)b  * IMHW);
    const float4* op = (const float4*)(o + (size_t)bp * IMHW);
    const float4* tm = (const float4*)(t + (size_t)bm * IMHW);
    const float4* tc = (const float4*)(t + (size_t)b  * IMHW);
    const float4* tp = (const float4*)(t + (size_t)bp * IMHW);
    __syncthreads();
    for (int q = tid; q < IMHW/4; q += 256) {          // [1,2,1] along batch
      float4 a4 = om[q], b4 = oc[q], c4 = op[q];
      float4 r;
      r.x = a4.x + 2.f*b4.x + c4.x; r.y = a4.y + 2.f*b4.y + c4.y;
      r.z = a4.z + 2.f*b4.z + c4.z; r.w = a4.w + 2.f*b4.w + c4.w;
      *(float4*)&sbx[q*4] = r;
      float4 d4 = tm[q], e4 = tc[q], f4 = tp[q];
      float4 s;
      s.x = d4.x + 2.f*e4.x + f4.x; s.y = d4.y + 2.f*e4.y + f4.y;
      s.z = d4.z + 2.f*e4.z + f4.z; s.w = d4.w + 2.f*e4.w + f4.w;
      *(float4*)&sby[q*4] = s;
    }
    __syncthreads();
    for (int px = tid; px < IMHW; px += 256) {
      int i = px / IMH, j = px - i*IMH;
      int jm = (j > 0) ? j-1 : 0, jp = (j < IMH-1) ? j+1 : IMH-1;
      int im = (i > 0) ? i-1 : 0, ip = (i < IMH-1) ? i+1 : IMH-1;
      float dxm = sbx[im*IMH + jp] - sbx[im*IMH + jm]; // [-1,0,1] along W
      float dxc = sbx[i *IMH + jp] - sbx[i *IMH + jm];
      float dxp = sbx[ip*IMH + jp] - sbx[ip*IMH + jm];
      float sx  = dxm + 2.f*dxc + dxp;                 // [1,2,1] along H
      float dym = sby[im*IMH + jp] - sby[im*IMH + jm];
      float dyc = sby[i *IMH + jp] - sby[i *IMH + jm];
      float dyp = sby[ip*IMH + jp] - sby[ip*IMH + jm];
      float sy  = dym + 2.f*dyc + dyp;
      a1 += sx; a2 += sy; a3 += sx*sx; a4 += sy*sy; a5 += sx*sy;
    }
  }
  a1 = blockReduceSum(a1, sRed, 256);
  a2 = blockReduceSum(a2, sRed, 256);
  a3 = blockReduceSum(a3, sRed, 256);
  a4 = blockReduceSum(a4, sRed, 256);
  a5 = blockReduceSum(a5, sRed, 256);
  if (tid == 0) {
    float* e = ws + EPI_BASE + 5*blockIdx.x;
    e[0] = a1; e[1] = a2; e[2] = a3; e[3] = a4; e[4] = a5;
  }
}

// ---------------- pass 5: combine ------------------------------------------
__global__ __launch_bounds__(256)
void k_final(float* __restrict__ ws, float* __restrict__ out)
{
  __shared__ float sA[256];
  int tid = threadIdx.x;
  float s = 0.f;
  for (int i = tid; i < NBLK; i += 256) s += ws[SSIM_BASE + i];
  float ssimSum = blockReduceSum(s, sA, 256);
  float eg[5];
  for (int q = 0; q < 5; ++q) {
    float v = 0.f;
    for (int i = tid; i < NBLK; i += 256) v += ws[EPI_BASE + 5*i + q];
    eg[q] = blockReduceSum(v, sA, 256);
  }
  if (tid == 0) {
    const double N = (double)NIMG * IMHW;
    double sx = eg[0], sy = eg[1], sxx = eg[2], syy = eg[3], sxy = eg[4];
    double num = sxy - sx*sy/N;
    double den = sqrt((sxx - sx*sx/N) * (syy - sy*sy/N));
    float epi = (float)(num / den);
    float mse = ws[0];
    float ssimMean = ssimSum / ((float)NIMG * (float)(NVALID*NVALID));
    float psnr = -10.f * log10f(mse);
    out[0] = 1.0f*mse + 0.5f*(1.f - ssimMean) + 0.1f*epi + 0.01f*psnr;
  }
}

// ---------------------------------------------------------------------------
extern "C" void kernel_launch(void* const* d_in, const int* in_sizes, int n_in,
                              void* d_out, int out_size, void* d_ws, size_t ws_size,
                              hipStream_t stream)
{
  (void)in_sizes; (void)n_in; (void)out_size; (void)ws_size;
  const float* o = (const float*)d_in[0];
  const float* t = (const float*)d_in[1];
  float* ws  = (float*)d_ws;
  float* out = (float*)d_out;

  k_stats<<<NBLK, 256, 0, stream>>>(o, t, ws);
  k_finalize_stats<<<1, 256, 0, stream>>>(ws);
  k_ssim<<<NBLK, 160, 0, stream>>>(o, t, ws);
  k_epi <<<NBLK, 256, 0, stream>>>(o, t, ws);
  k_final<<<1, 256, 0, stream>>>(ws, out);
}
